// Tacotron2Decoder_4672924418316
// MI455X (gfx1250) — matrix operand order repaired and checked by reading, not compiled, not run on hardware
//
#include <hip/hip_runtime.h>

#define B_    16
#define T_    384
#define S_    200
#define ENC_  512
#define MEL_  80
#define PREH_ 256
#define DECH_ 1024
#define ATT_  128
#define LOCC_ 32
#define LOCK_ 31

#define CDIV(a,b) (((a)+(b)-1)/(b))

typedef __attribute__((ext_vector_type(16))) _Float16 half16;
typedef __attribute__((ext_vector_type(8)))  _Float16 half8;
typedef __attribute__((ext_vector_type(8)))  float    float8;

__device__ __forceinline__ float sigf_(float x){ return 1.0f/(1.0f+__expf(-x)); }
__device__ __forceinline__ unsigned wanghash_(unsigned x){
  x ^= x>>16; x*=0x7feb352du; x^=x>>15; x*=0x846ca68bu; x^=x>>16; return x;
}

// ---------------------------------------------------------------- utility
__global__ void zero_ws_kernel(unsigned* p, size_t n){
  for (size_t i = (size_t)blockIdx.x*blockDim.x + threadIdx.x; i < n;
       i += (size_t)gridDim.x*blockDim.x) p[i] = 0u;
}

// dst[n*ldDst + colOff + k] = src[k*N + n]   (f32 [K,N] -> f16 [N,ldDst] transposed)
__global__ void pack_transpose_kernel(_Float16* dst, const float* src,
                                      int K, int N, int ldDst, int colOff){
  size_t idx = (size_t)blockIdx.x*blockDim.x + threadIdx.x;
  size_t total = (size_t)K*N;
  if (idx >= total) return;
  int k = (int)(idx % K); int n = (int)(idx / K);
  dst[(size_t)n*ldDst + colOff + k] = (_Float16)src[(size_t)k*N + n];
}

// dst[n*ldDst + k] = src[n*K + k]   (row-major convert+pad)
__global__ void pack_rows_kernel(_Float16* dst, const float* src, int K, int N, int ldDst){
  size_t idx = (size_t)blockIdx.x*blockDim.x + threadIdx.x;
  size_t total = (size_t)K*N;
  if (idx >= total) return;
  int k = (int)(idx % K); int n = (int)(idx / K);
  dst[(size_t)n*ldDst + k] = (_Float16)src[(size_t)n*K + k];
}

__global__ void f32_to_f16_kernel(_Float16* dst, const float* src, size_t n){
  for (size_t i = (size_t)blockIdx.x*blockDim.x + threadIdx.x; i < n;
       i += (size_t)gridDim.x*blockDim.x) dst[i] = (_Float16)src[i];
}

__global__ void add_vec_kernel(float* dst, const float* a, const float* b, int n){
  int i = blockIdx.x*blockDim.x + threadIdx.x;
  if (i < n) dst[i] = a[i] + b[i];
}

__global__ void dropout_to_f16_kernel(_Float16* dst, const float* src, size_t n, unsigned seed){
  for (size_t i = (size_t)blockIdx.x*blockDim.x + threadIdx.x; i < n;
       i += (size_t)gridDim.x*blockDim.x){
    unsigned h = wanghash_(seed ^ (unsigned)(i*2654435761u));
    dst[i] = (_Float16)((h & 0x80000000u) ? src[i]*2.0f : 0.0f);
  }
}

// ---------------------------------------------------------------- WMMA GEMM
// C[M,N](f32, ldC) = A[M,K](f16 row-major, ldA) * Bt[N,K](f16 row-major, ldB)^T + bias
// M,N multiples of 16; K multiple of 32. One wave per 16x16 output tile.
__global__ __launch_bounds__(256)
void wmma_gemm_kernel(float* __restrict__ C, const _Float16* __restrict__ A,
                      const _Float16* __restrict__ Bt,
                      int M, int N, int K, int ldA, int ldB, int ldC,
                      const float* __restrict__ bias, int relu){
  int wave   = blockIdx.x*(blockDim.x >> 5) + (threadIdx.x >> 5);
  int tilesN = N >> 4;
  int tiles  = (M >> 4) * tilesN;
  if (wave >= tiles) return;
  int tm = wave / tilesN, tn = wave % tilesN;
  int lane  = threadIdx.x & 31;
  int lrow  = lane & 15;
  int hi    = (lane >> 4) & 1;
  // A fragment (16x32 f16): lo lanes hold K {0..7,16..23}, hi lanes K {8..15,24..31}
  const _Float16* ap = A  + (size_t)(tm*16 + lrow)*ldA + (hi ? 8 : 0);
  // B fragment (32x16 f16): lane = column, lo lanes hold K {0..7,16..23}, hi lanes K {8..15,24..31}
  const _Float16* bp = Bt + (size_t)(tn*16 + lrow)*ldB + (hi ? 8 : 0);
  float8 acc = {};
  for (int k0 = 0; k0 < K; k0 += 32){
    __builtin_prefetch((const void*)(ap + 64), 0, 3);
    __builtin_prefetch((const void*)(bp + 64), 0, 3);
    half8 a0 = *(const half8*)(ap);
    half8 a1 = *(const half8*)(ap + 16);
    half8 b0 = *(const half8*)(bp);
    half8 b1 = *(const half8*)(bp + 16);
    half16 a = __builtin_shufflevector(a0, a1, 0,1,2,3,4,5,6,7,8,9,10,11,12,13,14,15);
    half16 b = __builtin_shufflevector(b0, b1, 0,1,2,3,4,5,6,7,8,9,10,11,12,13,14,15);
    acc = __builtin_amdgcn_wmma_f32_16x16x32_f16(false, a, false, b, (short)0, acc,
                                                 false, false);
    ap += 32; bp += 32;
  }
  int col   = tn*16 + lrow;
  int rbase = tm*16 + (hi ? 8 : 0);
  float bv  = bias ? bias[col] : 0.0f;
#pragma unroll
  for (int r = 0; r < 8; ++r){
    float v = acc[r] + bv;
    if (relu) v = v > 0.0f ? v : 0.0f;
    C[(size_t)(rbase + r)*ldC + col] = v;
  }
}

// ---------------------------------------------------------------- prenet input
__global__ void build_prenet_input_kernel(_Float16* A, const float* mel_targets){
  size_t idx = (size_t)blockIdx.x*blockDim.x + threadIdx.x;
  size_t total = (size_t)S_*B_*MEL_;
  if (idx >= total) return;
  int m = (int)(idx % MEL_);
  int b = (int)((idx / MEL_) % B_);
  int t = (int)(idx / (MEL_*B_));
  float v = (t == 0) ? 0.0f : mel_targets[((size_t)b*S_ + (t-1))*MEL_ + m];
  A[((size_t)t*B_ + b)*96 + m] = (_Float16)v;
}

// ---------------------------------------------------------------- attention step
__global__ __launch_bounds__(256)
void attn_step_kernel(const float* __restrict__ h2, float* __restrict__ prev_attn,
                      const float* __restrict__ memproj, const float* __restrict__ enc,
                      const float* __restrict__ Wq, const float* __restrict__ loc_w,
                      const float* __restrict__ loc_b, const float* __restrict__ Wloc,
                      const float* __restrict__ vvec, const float* __restrict__ abias,
                      const _Float16* __restrict__ pre_all, const float* __restrict__ h1,
                      float* __restrict__ attn_out, float* __restrict__ ctx,
                      _Float16* __restrict__ xin1, int t){
  __shared__ float qs[ATT_];
  __shared__ float es[T_];
  __shared__ float red[256];
  int b = blockIdx.x, tid = threadIdx.x;
  // q = h2 @ Wq + attn_bias
  if (tid < ATT_){
    float s = abias[tid];
    const float* hb = h2 + (size_t)b*DECH_;
    for (int k = 0; k < DECH_; ++k) s += hb[k]*Wq[(size_t)k*ATT_ + tid];
    qs[tid] = s;
  }
  __syncthreads();
  // energies
  for (int t2 = tid; t2 < T_; t2 += 256){
    float f[LOCC_];
#pragma unroll
    for (int c = 0; c < LOCC_; ++c) f[c] = loc_b[c];
    for (int j = 0; j < LOCK_; ++j){
      int tt = t2 + j - 15;
      if (tt >= 0 && tt < T_){
        float pa = prev_attn[(size_t)b*T_ + tt];
#pragma unroll
        for (int c = 0; c < LOCC_; ++c) f[c] += loc_w[c*LOCK_ + j]*pa;
      }
    }
    const float* mp = memproj + ((size_t)b*T_ + t2)*ATT_;
    float e = 0.0f;
    for (int n = 0; n < ATT_; ++n){
      float la = 0.0f;
#pragma unroll
      for (int c = 0; c < LOCC_; ++c) la += f[c]*Wloc[c*ATT_ + n];
      e += vvec[n]*tanhf(qs[n] + mp[n] + la);
    }
    es[t2] = e;
  }
  __syncthreads();
  // softmax over T
  float m = -1e30f;
  for (int t2 = tid; t2 < T_; t2 += 256) m = fmaxf(m, es[t2]);
  red[tid] = m; __syncthreads();
  for (int s = 128; s > 0; s >>= 1){ if (tid < s) red[tid] = fmaxf(red[tid], red[tid+s]); __syncthreads(); }
  m = red[0]; __syncthreads();
  float sum = 0.0f;
  for (int t2 = tid; t2 < T_; t2 += 256){ float ex = __expf(es[t2]-m); es[t2] = ex; sum += ex; }
  red[tid] = sum; __syncthreads();
  for (int s = 128; s > 0; s >>= 1){ if (tid < s) red[tid] += red[tid+s]; __syncthreads(); }
  float inv = 1.0f/red[0];
  __syncthreads();
  for (int t2 = tid; t2 < T_; t2 += 256){
    float a = es[t2]*inv; es[t2] = a;
    prev_attn[(size_t)b*T_ + t2] = a;
    attn_out[((size_t)b*S_ + t)*T_ + t2] = a;
  }
  __syncthreads();
  // ctx = attn @ enc ; pack xin1 = [pre(256), ctx(512), h1(1024)] f16
  for (int e0 = tid; e0 < ENC_; e0 += 256){
    float s = 0.0f;
    const float* eb = enc + (size_t)b*T_*ENC_ + e0;
    for (int t2 = 0; t2 < T_; ++t2) s += es[t2]*eb[(size_t)t2*ENC_];
    ctx[(size_t)b*ENC_ + e0] = s;
    xin1[(size_t)b*1792 + PREH_ + e0] = (_Float16)s;
  }
  for (int i = tid; i < PREH_; i += 256)
    xin1[(size_t)b*1792 + i] = pre_all[((size_t)t*B_ + b)*PREH_ + i];
  for (int i = tid; i < DECH_; i += 256)
    xin1[(size_t)b*1792 + 768 + i] = (_Float16)h1[(size_t)b*DECH_ + i];
}

// ---------------------------------------------------------------- LSTM pointwise
// gates [16,4096] -> h,c; also packs xcat[b*ld + 0..1023]=h_new, [ld offset 1024..]=extra
__global__ void lstm_pointwise_kernel(const float* __restrict__ gates, float* __restrict__ c,
                                      float* __restrict__ h, _Float16* __restrict__ xcat,
                                      int ld, const float* __restrict__ extra, int extra_n){
  int total = B_*DECH_ + B_*extra_n;
  for (int idx = blockIdx.x*blockDim.x + threadIdx.x; idx < total;
       idx += gridDim.x*blockDim.x){
    if (idx < B_*DECH_){
      int b = idx / DECH_, j = idx % DECH_;
      const float* g = gates + (size_t)b*4*DECH_;
      float gi = g[j], gf = g[DECH_+j], gg = g[2*DECH_+j], go = g[3*DECH_+j];
      float cn = sigf_(gf)*c[idx] + sigf_(gi)*tanhf(gg);
      c[idx] = cn;
      float hn = sigf_(go)*tanhf(cn);
      h[idx] = hn;
      xcat[(size_t)b*ld + j] = (_Float16)hn;
    } else {
      int k = idx - B_*DECH_;
      int b = k / extra_n, e = k % extra_n;
      xcat[(size_t)b*ld + DECH_ + e] = (_Float16)extra[(size_t)b*extra_n + e];
    }
  }
}

__global__ void stop_step_kernel(const float* __restrict__ h2, const float* __restrict__ ctx,
                                 const float* __restrict__ stop_w, const float* __restrict__ stop_b,
                                 float* __restrict__ out_stop, int t){
  __shared__ float red[256];
  int b = blockIdx.x, tid = threadIdx.x;
  float s = 0.0f;
  for (int j = tid; j < DECH_; j += 256) s += h2[(size_t)b*DECH_+j]*stop_w[j];
  for (int e = tid; e < ENC_; e += 256) s += ctx[(size_t)b*ENC_+e]*stop_w[DECH_+e];
  red[tid] = s; __syncthreads();
  for (int k = 128; k > 0; k >>= 1){ if (tid < k) red[tid] += red[tid+k]; __syncthreads(); }
  if (tid == 0) out_stop[(size_t)b*S_ + t] = red[0] + stop_b[0];
}

// ---------------------------------------------------------------- postnet
__global__ void transpose_mel_kernel(float* xbuf, const float* mel){ // [B,S,80] -> [B,80,S]
  size_t idx = (size_t)blockIdx.x*blockDim.x + threadIdx.x;
  if (idx >= (size_t)B_*MEL_*S_) return;
  int t = (int)(idx % S_);
  int c = (int)((idx / S_) % MEL_);
  int b = (int)(idx / ((size_t)S_*MEL_));
  xbuf[((size_t)b*MEL_ + c)*S_ + t] = mel[((size_t)b*S_ + t)*MEL_ + c];
}

__global__ void im2col_kernel(_Float16* A, const float* x, int Cin, int Kpad){
  size_t total = (size_t)B_*S_*Kpad;
  for (size_t idx = (size_t)blockIdx.x*blockDim.x + threadIdx.x; idx < total;
       idx += (size_t)gridDim.x*blockDim.x){
    int col = (int)(idx % Kpad);
    size_t r = idx / Kpad;
    int t = (int)(r % S_), b = (int)(r / S_);
    float v = 0.0f;
    if (col < Cin*5){
      int ci = col/5, j = col%5, tt = t + j - 2;
      if (tt >= 0 && tt < S_) v = x[((size_t)b*Cin + ci)*S_ + tt];
    }
    A[idx] = (_Float16)v;
  }
}

__global__ void bn_stats_kernel(const float* __restrict__ y, float* __restrict__ stats, int Co){
  __shared__ float r1[256], r2[256];
  int c = blockIdx.x, tid = threadIdx.x;
  float s = 0.0f, sq = 0.0f;
  for (int r = tid; r < B_*S_; r += 256){
    float v = y[(size_t)r*Co + c];
    s += v; sq += v*v;
  }
  r1[tid] = s; r2[tid] = sq; __syncthreads();
  for (int k = 128; k > 0; k >>= 1){
    if (tid < k){ r1[tid] += r1[tid+k]; r2[tid] += r2[tid+k]; }
    __syncthreads();
  }
  if (tid == 0){
    float mean = r1[0]/(float)(B_*S_);
    float var  = r2[0]/(float)(B_*S_) - mean*mean;
    stats[2*c]   = mean;
    stats[2*c+1] = rsqrtf(var + 1e-5f);
  }
}

__global__ void bn_apply_kernel(float* __restrict__ xout, const float* __restrict__ y,
                                const float* __restrict__ stats, const float* __restrict__ g,
                                const float* __restrict__ be, int Co){
  size_t idx = (size_t)blockIdx.x*blockDim.x + threadIdx.x;
  if (idx >= (size_t)B_*S_*Co) return;
  int c = (int)(idx % Co);
  size_t r = idx / Co;
  int t = (int)(r % S_), b = (int)(r / S_);
  float v = (y[(size_t)r*Co + c] - stats[2*c])*stats[2*c+1]*g[c] + be[c];
  xout[((size_t)b*Co + c)*S_ + t] = tanhf(v);
}

__global__ void bn_final_kernel(float* __restrict__ mel_post, const float* __restrict__ mel_out,
                                const float* __restrict__ y, const float* __restrict__ stats,
                                const float* __restrict__ g, const float* __restrict__ be){
  size_t idx = (size_t)blockIdx.x*blockDim.x + threadIdx.x;
  if (idx >= (size_t)B_*S_*MEL_) return;
  int c = (int)(idx % MEL_);
  size_t r = idx / MEL_;
  float v = (y[(size_t)r*MEL_ + c] - stats[2*c])*stats[2*c+1]*g[c] + be[c];
  mel_post[idx] = mel_out[idx] + v;
}

// ---------------------------------------------------------------- host
extern "C" void kernel_launch(void* const* d_in, const int* in_sizes, int n_in,
                              void* d_out, int out_size, void* d_ws, size_t ws_size,
                              hipStream_t stream){
  (void)in_sizes; (void)n_in; (void)out_size;
  const float* p_enc     = (const float*)d_in[0];
  const float* p_melT    = (const float*)d_in[1];
  const float* p_pre_w1  = (const float*)d_in[3];
  const float* p_pre_b1  = (const float*)d_in[4];
  const float* p_pre_w2  = (const float*)d_in[5];
  const float* p_pre_b2  = (const float*)d_in[6];
  const float* p_attn_qw = (const float*)d_in[7];
  const float* p_attn_mw = (const float*)d_in[8];
  const float* p_attn_lw = (const float*)d_in[9];
  const float* p_loc_w   = (const float*)d_in[10];
  const float* p_loc_b   = (const float*)d_in[11];
  const float* p_attn_v  = (const float*)d_in[12];
  const float* p_attn_bi = (const float*)d_in[13];
  const float* p_l1_wih  = (const float*)d_in[14];
  const float* p_l1_whh  = (const float*)d_in[15];
  const float* p_l1_bih  = (const float*)d_in[16];
  const float* p_l1_bhh  = (const float*)d_in[17];
  const float* p_l2_wih  = (const float*)d_in[18];
  const float* p_l2_whh  = (const float*)d_in[19];
  const float* p_l2_bih  = (const float*)d_in[20];
  const float* p_l2_bhh  = (const float*)d_in[21];
  const float* p_mel_w   = (const float*)d_in[22];
  const float* p_mel_b   = (const float*)d_in[23];
  const float* p_stop_w  = (const float*)d_in[24];
  const float* p_stop_b  = (const float*)d_in[25];
  const float *p_pn_w[5], *p_pn_b[5], *p_pn_g[5], *p_pn_be[5];
  for (int i = 0; i < 5; ++i){
    p_pn_w[i]  = (const float*)d_in[26 + 4*i];
    p_pn_b[i]  = (const float*)d_in[27 + 4*i];
    p_pn_g[i]  = (const float*)d_in[28 + 4*i];
    p_pn_be[i] = (const float*)d_in[29 + 4*i];
  }

  // workspace bump allocator --------------------------------------------
  size_t off = 0;
  auto WA = [&](size_t elems, size_t esz)->void*{
    off = (off + 255) & ~(size_t)255;
    void* p = (char*)d_ws + off;
    off += elems*esz;
    return p;
  };
  // --- dead-after-steps region (aliased later by postnet scratch) ---
  _Float16* W1t    = (_Float16*)WA((size_t)4096*1792, 2);
  _Float16* W2t    = (_Float16*)WA((size_t)4096*2048, 2);
  _Float16* melWt  = (_Float16*)WA((size_t)80*1536, 2);
  _Float16* preW1t = (_Float16*)WA((size_t)256*96, 2);
  _Float16* preW2t = (_Float16*)WA((size_t)256*256, 2);
  _Float16* attnMWt= (_Float16*)WA((size_t)128*512, 2);
  _Float16* enc16  = (_Float16*)WA((size_t)6144*512, 2);
  _Float16* Apre   = (_Float16*)WA((size_t)3200*96, 2);
  float*    preTmp = (float*)   WA((size_t)3200*256, 4);
  _Float16* preA2  = (_Float16*)WA((size_t)3200*256, 2);
  _Float16* preAll = (_Float16*)WA((size_t)3200*256, 2);
  float*    memproj= (float*)   WA((size_t)6144*128, 4);
  float*    gates1 = (float*)   WA((size_t)16*4096, 4);
  float*    gates2 = (float*)   WA((size_t)16*4096, 4);
  _Float16* xin1   = (_Float16*)WA((size_t)16*1792, 2);
  _Float16* xin2   = (_Float16*)WA((size_t)16*2048, 2);
  _Float16* pibuf  = (_Float16*)WA((size_t)16*1536, 2);
  // --- persistent region ---
  _Float16* pnWt[5];
  pnWt[0] = (_Float16*)WA((size_t)512*416, 2);
  pnWt[1] = (_Float16*)WA((size_t)512*2560, 2);
  pnWt[2] = (_Float16*)WA((size_t)512*2560, 2);
  pnWt[3] = (_Float16*)WA((size_t)512*2560, 2);
  pnWt[4] = (_Float16*)WA((size_t)80*2560, 2);
  float* bias1 = (float*)WA(4096, 4);
  float* bias2 = (float*)WA(4096, 4);
  float* h1 = (float*)WA((size_t)16*1024, 4);
  float* c1 = (float*)WA((size_t)16*1024, 4);
  float* h2 = (float*)WA((size_t)16*1024, 4);
  float* c2 = (float*)WA((size_t)16*1024, 4);
  float* prevAttn = (float*)WA((size_t)16*384, 4);
  float* ctx = (float*)WA((size_t)16*512, 4);
  float* stats = (float*)WA(1024, 4);
  // --- postnet arena: aliases the dead region starting at ws offset 0 ---
  size_t poff = 0;
  auto PA = [&](size_t elems, size_t esz)->void*{
    poff = (poff + 255) & ~(size_t)255;
    void* p = (char*)d_ws + poff;
    poff += elems*esz;
    return p;
  };
  _Float16* Apn = (_Float16*)PA((size_t)3200*2560, 2);
  float*    ypn = (float*)   PA((size_t)3200*512, 4);
  float*    xb0 = (float*)   PA((size_t)16*512*200, 4);
  float*    xb1 = (float*)   PA((size_t)16*512*200, 4);

  float* out          = (float*)d_out;
  float* out_melpost  = out;                    // [16,200,80]
  float* out_mel      = out + 256000;           // [16,200,80]
  float* out_stop     = out + 512000;           // [16,200,1]
  float* out_attn     = out + 515200;           // [16,200,384]

  // 0) zero the whole workspace: zero-inits states AND all K-pads.
  zero_ws_kernel<<<2048, 256, 0, stream>>>((unsigned*)d_ws, ws_size/4);

  // 1) pack weights (f32 -> f16, transposed to [N,K]) -------------------
  pack_transpose_kernel<<<CDIV(80*256,256),256,0,stream>>>(preW1t, p_pre_w1, 80, 256, 96, 0);
  pack_transpose_kernel<<<CDIV(256*256,256),256,0,stream>>>(preW2t, p_pre_w2, 256, 256, 256, 0);
  pack_transpose_kernel<<<CDIV(512*128,256),256,0,stream>>>(attnMWt, p_attn_mw, 512, 128, 512, 0);
  pack_transpose_kernel<<<CDIV(768*4096,256),256,0,stream>>>(W1t, p_l1_wih, 768, 4096, 1792, 0);
  pack_transpose_kernel<<<CDIV(1024*4096,256),256,0,stream>>>(W1t, p_l1_whh, 1024, 4096, 1792, 768);
  pack_transpose_kernel<<<CDIV(1024*4096,256),256,0,stream>>>(W2t, p_l2_wih, 1024, 4096, 2048, 0);
  pack_transpose_kernel<<<CDIV(1024*4096,256),256,0,stream>>>(W2t, p_l2_whh, 1024, 4096, 2048, 1024);
  pack_transpose_kernel<<<CDIV(1536*80,256),256,0,stream>>>(melWt, p_mel_w, 1536, 80, 1536, 0);
  pack_rows_kernel<<<CDIV(512*400,256),256,0,stream>>>(pnWt[0], p_pn_w[0], 400, 512, 416);
  for (int i = 1; i <= 3; ++i)
    pack_rows_kernel<<<CDIV(512*2560,256),256,0,stream>>>(pnWt[i], p_pn_w[i], 2560, 512, 2560);
  pack_rows_kernel<<<CDIV(80*2560,256),256,0,stream>>>(pnWt[4], p_pn_w[4], 2560, 80, 2560);
  add_vec_kernel<<<16,256,0,stream>>>(bias1, p_l1_bih, p_l1_bhh, 4096);
  add_vec_kernel<<<16,256,0,stream>>>(bias2, p_l2_bih, p_l2_bhh, 4096);

  // 2) one-time: mem_proj = enc @ attn_m_w  (WMMA, M=6144,K=512,N=128)
  f32_to_f16_kernel<<<4096,256,0,stream>>>(enc16, p_enc, (size_t)6144*512);
  wmma_gemm_kernel<<<CDIV(384*8,8),256,0,stream>>>(memproj, enc16, attnMWt,
      6144, 128, 512, 512, 512, 128, nullptr, 0);

  // 3) one-time: prenet for all 200 teacher-forced steps (M=3200)
  build_prenet_input_kernel<<<CDIV(200*16*80,256),256,0,stream>>>(Apre, p_melT);
  wmma_gemm_kernel<<<CDIV(3200,8),256,0,stream>>>(preTmp, Apre, preW1t,
      3200, 256, 96, 96, 96, 256, p_pre_b1, 1);
  dropout_to_f16_kernel<<<3200,256,0,stream>>>(preA2, preTmp, (size_t)3200*256, 0x9E3779B9u);
  wmma_gemm_kernel<<<CDIV(3200,8),256,0,stream>>>(preTmp, preA2, preW2t,
      3200, 256, 256, 256, 256, 256, p_pre_b2, 1);
  dropout_to_f16_kernel<<<3200,256,0,stream>>>(preAll, preTmp, (size_t)3200*256, 0x85EBCA6Bu);

  // 4) 200 sequential decoder steps -------------------------------------
  for (int t = 0; t < S_; ++t){
    attn_step_kernel<<<16,256,0,stream>>>(h2, prevAttn, memproj, p_enc, p_attn_qw,
        p_loc_w, p_loc_b, p_attn_lw, p_attn_v, p_attn_bi, preAll, h1,
        out_attn, ctx, xin1, t);
    wmma_gemm_kernel<<<32,256,0,stream>>>(gates1, xin1, W1t,
        16, 4096, 1792, 1792, 1792, 4096, bias1, 0);
    lstm_pointwise_kernel<<<128,256,0,stream>>>(gates1, c1, h1, xin2, 2048, h2, 1024);
    wmma_gemm_kernel<<<32,256,0,stream>>>(gates2, xin2, W2t,
        16, 4096, 2048, 2048, 2048, 4096, bias2, 0);
    lstm_pointwise_kernel<<<96,256,0,stream>>>(gates2, c2, h2, pibuf, 1536, ctx, 512);
    wmma_gemm_kernel<<<1,256,0,stream>>>(out_mel + t*MEL_, pibuf, melWt,
        16, 80, 1536, 1536, 1536, S_*MEL_, p_mel_b, 0);
    stop_step_kernel<<<16,256,0,stream>>>(h2, ctx, p_stop_w, p_stop_b, out_stop, t);
  }

  // 5) postnet (im2col -> WMMA GEMM -> BN) ------------------------------
  transpose_mel_kernel<<<CDIV(256000,256),256,0,stream>>>(xb0, out_mel);
  float* xcur = xb0; float* xnext = xb1;
  for (int i = 0; i < 5; ++i){
    int Cin = (i == 0) ? MEL_ : 512;
    int Co  = (i == 4) ? MEL_ : 512;
    int Kp  = (i == 0) ? 416 : 2560;
    im2col_kernel<<<4096,256,0,stream>>>(Apn, xcur, Cin, Kp);
    int tiles = (3200/16)*(Co/16);
    wmma_gemm_kernel<<<CDIV(tiles,8),256,0,stream>>>(ypn, Apn, pnWt[i],
        3200, Co, Kp, Kp, Kp, Co, p_pn_b[i], 0);
    bn_stats_kernel<<<Co,256,0,stream>>>(ypn, stats, Co);
    if (i < 4){
      bn_apply_kernel<<<CDIV(3200*Co,256),256,0,stream>>>(xnext, ypn, stats,
          p_pn_g[i], p_pn_be[i], Co);
      float* tmp = xcur; xcur = xnext; xnext = tmp;
    } else {
      bn_final_kernel<<<CDIV(256000,256),256,0,stream>>>(out_melpost, out_mel, ypn,
          stats, p_pn_g[4], p_pn_be[4]);
    }
  }
}
